// EfficientCrossModalAttention_85048942395451
// MI455X (gfx1250) — matrix operand order repaired and checked by reading, not compiled, not run on hardware
//
#include <hip/hip_runtime.h>

typedef __attribute__((ext_vector_type(16))) _Float16 v16h;
typedef __attribute__((ext_vector_type(8)))  _Float16 v8h;
typedef __attribute__((ext_vector_type(8)))  float    v8f;
typedef __attribute__((ext_vector_type(4)))  float    v4f;

#define BATCH 4
#define CCH   256
#define NTOK  4096
#define DDIM  32

union AV { v16h v; v8h h[2]; };

__device__ __forceinline__ v8f wmma16(v16h a, v16h b, v8f c) {
  // D = A(16x32 f16) * B(32x16 f16) + C(16x16 f32)
  return __builtin_amdgcn_wmma_f32_16x16x32_f16(false, a, false, b, (short)0, c,
                                                false, false);
}
// reduce across the two 16-lane halves of a wave32 (SWAPX16 swizzle)
__device__ __forceinline__ float xor16(float v) {
  return __int_as_float(__builtin_amdgcn_ds_swizzle(__float_as_int(v), 0x401f));
}
// broadcast value from lane `src` (0..31)
__device__ __forceinline__ float bcast(float v, int src) {
  return __int_as_float(__builtin_amdgcn_ds_bpermute(src << 2, __float_as_int(v)));
}
__device__ __forceinline__ float fexp2(float x) {
  return __builtin_amdgcn_exp2f(x);  // raw v_exp_f32, scores pre-scaled by log2e
}

// ---------------------------------------------------------------------------
// Kernel 0: one-shot f32 -> f16 conversion of the weight matrices so the
// GEMM kernels load B-tiles as b128 with zero per-wave conversion VALU.
// ---------------------------------------------------------------------------
__global__ __launch_bounds__(256) void cvtw_kernel(
    const float* __restrict__ w_qkv, const float* __restrict__ w_proj,
    _Float16* __restrict__ wq16, _Float16* __restrict__ wp16) {
  const int i = blockIdx.x * 256 + threadIdx.x;
  if (i < 96 * CCH)  wq16[i] = (_Float16)w_qkv[i];
  if (i < CCH * DDIM) wp16[i] = (_Float16)w_proj[i];
}

// ---------------------------------------------------------------------------
// Kernel 1: QKV = tok @ w_qkv^T + b_qkv   (tok[b][n][c] = x[b][c][n])
// q scaled by log2(e)/sqrt(D) and stored f16 [b][n][d]; k f16 [b][n][d];
// v stored transposed f16 vt[b][d][n].
// One wave per 16-token tile; 6 j-tiles x 8 K=32 chunks of WMMA.
// ---------------------------------------------------------------------------
__global__ __launch_bounds__(128) void qkv_kernel(
    const float* __restrict__ x, const _Float16* __restrict__ wq16,
    const float* __restrict__ b_qkv,
    _Float16* __restrict__ qbuf, _Float16* __restrict__ kbuf,
    _Float16* __restrict__ vt) {
  const int wid  = threadIdx.x >> 5;
  const int lane = threadIdx.x & 31;
  const int tile = blockIdx.x * 4 + wid;       // 0..1023
  const int b    = tile >> 8;
  const int n0   = (tile & 255) << 4;
  const int half = lane >> 4, lq = lane & 15;

  v8f acc[6];
#pragma unroll
  for (int j = 0; j < 6; ++j)
    acc[j] = v8f{0.f, 0.f, 0.f, 0.f, 0.f, 0.f, 0.f, 0.f};

  const float* xrow = x + (size_t)b * CCH * NTOK + n0 + lq;  // + c*NTOK
#pragma unroll 1
  for (int kc = 0; kc < 8; ++kc) {
    const int cbase = kc * 32;
    AV a;  // A-tile: row = token lq, 32 channels of this chunk
#pragma unroll
    for (int h = 0; h < 8; ++h) {
      a.v[h]     = (_Float16)xrow[(size_t)(cbase + 8 * half + h) * NTOK];
      a.v[h + 8] = (_Float16)xrow[(size_t)(cbase + 16 + 8 * half + h) * NTOK];
    }
#pragma unroll
    for (int jt = 0; jt < 6; ++jt) {
      // B-tile: b[h] = w_qkv[jt*16+lq][cbase + 8*half + h], b[h+8] = ...[cbase + 16 + 8*half + h]  (pre-converted)
      const _Float16* wrow = wq16 + (size_t)(jt * 16 + lq) * CCH + cbase + 8 * half;
      AV bm;
      bm.h[0] = *(const v8h*)wrow;
      bm.h[1] = *(const v8h*)(wrow + 16);
      acc[jt] = wmma16(a.v, bm.v, acc[jt]);
    }
  }
  const float qscale = 0.17677669529663687f * 1.4426950408889634f;  // log2e/sqrt(32)
#pragma unroll
  for (int jt = 0; jt < 6; ++jt) {
    const int j = jt * 16 + lq;
    const float bias = b_qkv[j];
#pragma unroll
    for (int r = 0; r < 8; ++r) {
      const int n = n0 + r + 8 * half;  // D-layout row
      float v = acc[jt][r] + bias;
      if (jt < 2)
        qbuf[((size_t)b * NTOK + n) * DDIM + j] = (_Float16)(v * qscale);
      else if (jt < 4)
        kbuf[((size_t)b * NTOK + n) * DDIM + (j - 32)] = (_Float16)v;
      else
        vt[((size_t)b * DDIM + (j - 64)) * NTOK + n] = (_Float16)v;
    }
  }
}

// ---------------------------------------------------------------------------
// Kernel 2: flash attention. One wave per 16-query tile, loop over 128
// chunks of 32 keys. S^T = K_tile @ Q^T so exp(S) lands directly in the
// A-layout needed by the P@V WMMA (no cross-lane fixup). Scores are in the
// log2 domain (scale folded into q), so exp = raw v_exp_f32.
// ---------------------------------------------------------------------------
__global__ __launch_bounds__(128) void attn_kernel(
    const _Float16* __restrict__ qbuf, const _Float16* __restrict__ kbuf,
    const _Float16* __restrict__ vt, _Float16* __restrict__ obuf) {
  const int wid  = threadIdx.x >> 5;
  const int lane = threadIdx.x & 31;
  const int tile = blockIdx.x * 4 + wid;
  const int b    = tile >> 8;
  const int n0   = (tile & 255) << 4;
  const int half = lane >> 4, lq = lane & 15;

  // B = Q^T: b[h] = Q[q=lq][d = h + 8*half], b[h+8] = Q[q=lq][d = 16 + h + 8*half]
  const _Float16* qrow = qbuf + ((size_t)b * NTOK + n0 + lq) * DDIM + 8 * half;
  AV bq;
  bq.h[0] = *(const v8h*)qrow;
  bq.h[1] = *(const v8h*)(qrow + 16);

  const _Float16* kbase = kbuf + ((size_t)b * NTOK + lq) * DDIM + 8 * half;
  const _Float16* vbase = vt + ((size_t)b * DDIM + lq) * NTOK + 8 * half;

  v8f oc0 = {0.f, 0.f, 0.f, 0.f, 0.f, 0.f, 0.f, 0.f};
  v8f oc1 = oc0;
  float m = -1e30f, l = 0.f;

#pragma unroll 1
  for (int kc = 0; kc < NTOK / 32; ++kc) {
    const int c0 = kc * 32;
    // A = K tile: row = key c0+lq (+16 for ak1), a[h] per CDNA5 16-bit A layout
    const _Float16* kr0 = kbase + (size_t)c0 * DDIM;
    AV ak0, ak1;
    ak0.h[0] = *(const v8h*)kr0;
    ak0.h[1] = *(const v8h*)(kr0 + 16);
    const _Float16* kr1 = kr0 + 16 * DDIM;
    ak1.h[0] = *(const v8h*)kr1;
    ak1.h[1] = *(const v8h*)(kr1 + 16);

    if (kc + 1 < NTOK / 32) {  // cover next chunk's latency
      __builtin_prefetch(kr0 + 32 * DDIM, 0, 1);
      __builtin_prefetch(vbase + c0 + 32, 0, 1);
    }

    v8f s0 = {0.f, 0.f, 0.f, 0.f, 0.f, 0.f, 0.f, 0.f};
    v8f s1 = s0;
    s0 = wmma16(ak0.v, bq.v, s0);  // S^T keys c0..c0+15
    s1 = wmma16(ak1.v, bq.v, s1);  // S^T keys c0+16..c0+31

    // online softmax: this lane owns query q = lq
    float tmax = fmaxf(s0[0], s1[0]);
#pragma unroll
    for (int r = 1; r < 8; ++r) tmax = fmaxf(tmax, fmaxf(s0[r], s1[r]));
    tmax = fmaxf(tmax, xor16(tmax));

    // Rescale only when some lane's running max grew (wave-uniform branch:
    // ds_bpermute must not run under divergent EXEC).
    if (__any(tmax > m)) {
      const float mnew  = fmaxf(m, tmax);
      const float alpha = fexp2(m - mnew);  // ==1 for lanes whose max held
      l *= alpha;
#pragma unroll
      for (int r = 0; r < 8; ++r) {
        const float ar = bcast(alpha, r + 8 * half);  // row q' = r + 8*half
        oc0[r] *= ar;
        oc1[r] *= ar;
      }
      m = mnew;
    }

    AV ap;  // P in A-layout (f16)
    float psum = 0.f;
#pragma unroll
    for (int r = 0; r < 8; ++r) {
      float e0 = fexp2(s0[r] - m);
      float e1 = fexp2(s1[r] - m);
      psum += e0 + e1;
      ap.v[r]     = (_Float16)e0;
      ap.v[r + 8] = (_Float16)e1;
    }
    l += psum + xor16(psum);

    // B = V chunk columns: b[h] = Vt[d][c0 + 8*half + h], b[h+8] = Vt[d][c0 + 16 + 8*half + h]
    const _Float16* vr0 = vbase + c0;
    AV bv0;
    bv0.h[0] = *(const v8h*)vr0;
    bv0.h[1] = *(const v8h*)(vr0 + 16);
    const _Float16* vr1 = vr0 + (size_t)16 * NTOK;
    AV bv1;
    bv1.h[0] = *(const v8h*)vr1;
    bv1.h[1] = *(const v8h*)(vr1 + 16);

    oc0 = wmma16(ap.v, bv0.v, oc0);  // O[:, 0:16]
    oc1 = wmma16(ap.v, bv1.v, oc1);  // O[:, 16:32]
  }

  const float linv = 1.f / l;
#pragma unroll
  for (int r = 0; r < 8; ++r) {
    const float lr = bcast(linv, r + 8 * half);
    oc0[r] *= lr;
    oc1[r] *= lr;
  }
  _Float16* orow = obuf + ((size_t)b * NTOK + n0) * DDIM;
#pragma unroll
  for (int r = 0; r < 8; ++r) {
    const int nl = r + 8 * half;
    orow[(size_t)nl * DDIM + lq]      = (_Float16)oc0[r];
    orow[(size_t)nl * DDIM + 16 + lq] = (_Float16)oc1[r];
  }
}

// ---------------------------------------------------------------------------
// Kernel 3: proj = O @ w_proj^T + b_proj, then out[b][c][n] = proj + x.
// WMMA per 16x16 c-tile; LDS transpose for coalesced stores.
// ---------------------------------------------------------------------------
__global__ __launch_bounds__(128) void proj_kernel(
    const _Float16* __restrict__ obuf, const _Float16* __restrict__ wp16,
    const float* __restrict__ b_proj, const float* __restrict__ x,
    float* __restrict__ out) {
  __shared__ float tl[4][256];
  const int wid  = threadIdx.x >> 5;
  const int lane = threadIdx.x & 31;
  const int tile = blockIdx.x * 4 + wid;
  const int b    = tile >> 8;
  const int n0   = (tile & 255) << 4;
  const int half = lane >> 4, lq = lane & 15;

  const _Float16* orow = obuf + ((size_t)b * NTOK + n0 + lq) * DDIM;
  AV a;  // A = O tile 16x32
  a.h[0] = *(const v8h*)(orow + 8 * half);
  a.h[1] = *(const v8h*)(orow + 16 + 8 * half);

#pragma unroll 1
  for (int ct = 0; ct < 16; ++ct) {
    const int c0 = ct * 16;
    // B[d][c] = w_proj[c0+lq][d]  (pre-converted f16)
    const _Float16* wrow = wp16 + (size_t)(c0 + lq) * DDIM + 8 * half;
    AV bm;
    bm.h[0] = *(const v8h*)wrow;
    bm.h[1] = *(const v8h*)(wrow + 16);

    v8f d = {0.f, 0.f, 0.f, 0.f, 0.f, 0.f, 0.f, 0.f};
    d = wmma16(a.v, bm.v, d);
    const float bias = b_proj[c0 + lq];
#pragma unroll
    for (int r = 0; r < 8; ++r)
      tl[wid][lq * 16 + r + 8 * half] = d[r] + bias;  // [c_local][n_local]
    asm volatile("s_wait_dscnt 0x0" ::: "memory");
    const size_t outbase = ((size_t)b * CCH + c0) * NTOK + n0;
#pragma unroll
    for (int i = 0; i < 8; ++i) {
      const int cl = 2 * i + half;
      const float v = tl[wid][cl * 16 + lq];
      const size_t idx = outbase + (size_t)cl * NTOK + lq;
      out[idx] = v + x[idx];  // coalesced 64B per half-wave
    }
    asm volatile("s_wait_dscnt 0x0" ::: "memory");  // WAR fence vs next iter
  }
}

extern "C" void kernel_launch(void* const* d_in, const int* in_sizes, int n_in,
                              void* d_out, int out_size, void* d_ws,
                              size_t ws_size, hipStream_t stream) {
  const float* x      = (const float*)d_in[0];
  const float* w_qkv  = (const float*)d_in[1];
  const float* b_qkv  = (const float*)d_in[2];
  const float* w_proj = (const float*)d_in[3];
  const float* b_proj = (const float*)d_in[4];
  float* out = (float*)d_out;

  char* ws = (char*)d_ws;
  const size_t seg = (size_t)BATCH * NTOK * DDIM * sizeof(_Float16);  // 1 MB
  _Float16* qbuf = (_Float16*)(ws);
  _Float16* kbuf = (_Float16*)(ws + seg);
  _Float16* vt   = (_Float16*)(ws + 2 * seg);
  _Float16* obuf = (_Float16*)(ws + 3 * seg);
  _Float16* wq16 = (_Float16*)(ws + 4 * seg);                       // 48 KB
  _Float16* wp16 = (_Float16*)(ws + 4 * seg + 96 * CCH * sizeof(_Float16));

  cvtw_kernel<<<(96 * CCH + 255) / 256, 256, 0, stream>>>(w_qkv, w_proj, wq16,
                                                          wp16);
  const int tiles = BATCH * NTOK / 16;  // 1024 waves, 4 waves/block
  qkv_kernel<<<tiles / 4, 128, 0, stream>>>(x, wq16, b_qkv, qbuf, kbuf, vt);
  attn_kernel<<<tiles / 4, 128, 0, stream>>>(qbuf, kbuf, vt, obuf);
  proj_kernel<<<tiles / 4, 128, 0, stream>>>(obuf, wp16, b_proj, x, out);
}
